// NSA_45724221833311
// MI455X (gfx1250) — matrix operand order repaired and checked by reading, not compiled, not run on hardware
//
#include <hip/hip_runtime.h>

typedef _Float16 half_t;
typedef __attribute__((ext_vector_type(16))) _Float16 v16h;
typedef __attribute__((ext_vector_type(8)))  _Float16 v8h;
typedef __attribute__((ext_vector_type(8)))  float    v8f;

#define T_LEN 16384
#define D_DIM 128
#define BS    64
#define NB    256            // T_LEN / BS
#define WAVES 4
#define MROWS (WAVES * 16)   // 64 Q rows per workgroup

// ---------------------------------------------------------------------------
// Kernel 1: block mean-pool compression.
//   comp_K16 : [B][NB][D]  row-major  (so QK^T B-fragments are contiguous)
//   comp_VT16: [B][D][NB]  transposed (so PV   B-fragments are contiguous)
// One block per (b, n); thread d handles one feature column (coalesced).
// ---------------------------------------------------------------------------
__global__ __launch_bounds__(128)
void nsa_compress(const float* __restrict__ K, const float* __restrict__ V,
                  half_t* __restrict__ cK, half_t* __restrict__ cVT)
{
    const int bn = blockIdx.x;
    const int b  = bn / NB;
    const int n  = bn % NB;
    const int d  = threadIdx.x;              // 0..127

    const size_t base = ((size_t)b * T_LEN + (size_t)n * BS) * D_DIM + d;
    const float* kp = K + base;
    const float* vp = V + base;

    float sk = 0.f, sv = 0.f;
    #pragma unroll 4
    for (int j = 0; j < BS; ++j) {
        sk += kp[(size_t)j * D_DIM];
        sv += vp[(size_t)j * D_DIM];
    }
    const float inv = 1.0f / (float)BS;
    cK [((size_t)b * NB + n) * D_DIM + d] = (half_t)(sk * inv);
    cVT[((size_t)b * D_DIM + d) * NB + n] = (half_t)(sv * inv);
}

// ---------------------------------------------------------------------------
// Kernel 2: scores -> softmax -> P.V -> gate, all WMMA f32<-f16.
// 4 waves x 16 Q rows. comp_K/comp_V stay L2-resident (128 KB per batch).
// ---------------------------------------------------------------------------
__global__ __launch_bounds__(128)
void nsa_attn(const float* __restrict__ Q,
              const half_t* __restrict__ cK,   // [B][NB][D]
              const half_t* __restrict__ cVT,  // [B][D][NB]
              const float* __restrict__ gate,  // [B][T]
              float* __restrict__ out)         // [B][T][D]
{
    __shared__ __align__(32) half_t Plds[WAVES * 16 * NB];  // 32 KB

    const int tid  = threadIdx.x;
    const int wave = tid >> 5;
    const int lane = tid & 31;
    const int l16  = lane & 15;
    const int hi   = lane >> 4;          // 0: lanes 0-15, 1: lanes 16-31

    const int tilesPerB = T_LEN / MROWS;
    const int b  = blockIdx.x / tilesPerB;
    const int t0 = (blockIdx.x % tilesPerB) * MROWS + wave * 16;  // wave's row base

    // ---- Q A-fragments: 4 chunks of K=32, f32 -> f16 with 1/sqrt(D) folded in
    const float scale = 0.08838834764831845f;  // 128^-0.5
    const int   mrow  = t0 + l16;              // A layout: row = lane%16 (both halves)
    const float* qrow = Q + ((size_t)b * T_LEN + mrow) * D_DIM;
    v16h aq[4];
    #pragma unroll
    for (int kc = 0; kc < 4; ++kc) {
        const int k0 = kc * 32 + hi * 16;      // lanes<16: K 0-15, else 16-31
        #pragma unroll
        for (int i = 0; i < 8; ++i) {
            aq[kc][i]     = (half_t)(qrow[k0 + i]      * scale);
            aq[kc][i + 8] = (half_t)(qrow[k0 + 8 + i] * scale);
        }
    }

    // ---- scores: 16 n-tiles of 16 blocks each, K = D = 128
    v8f acc[16];
    const half_t* cKb = cK + (size_t)b * NB * D_DIM;
    #pragma unroll
    for (int nt = 0; nt < 16; ++nt) {
        v8f c = {};
        const int ncol = nt * 16 + l16;                       // B col = lane%16
        const half_t* krow = cKb + (size_t)ncol * D_DIM + hi * 16;
        #pragma unroll
        for (int kc = 0; kc < 4; ++kc) {
            v16h bk = *(const v16h*)(krow + kc * 32);         // 32B contiguous
            c = __builtin_amdgcn_wmma_f32_16x16x32_f16(false, aq[kc],
                                                       false, bk,
                                                       (short)0, c, false, false);
        }
        acc[nt] = c;
    }

    // ---- softmax over 256 scores/row (16 lanes in a half x 16 accumulators)
    #pragma unroll
    for (int r = 0; r < 8; ++r) {
        float m = acc[0][r];
        #pragma unroll
        for (int nt = 1; nt < 16; ++nt) m = fmaxf(m, acc[nt][r]);
        #pragma unroll
        for (int off = 1; off < 16; off <<= 1) m = fmaxf(m, __shfl_xor(m, off, 16));

        float s = 0.f;
        #pragma unroll
        for (int nt = 0; nt < 16; ++nt) {
            float e = __expf(acc[nt][r] - m);
            acc[nt][r] = e;
            s += e;
        }
        #pragma unroll
        for (int off = 1; off < 16; off <<= 1) s += __shfl_xor(s, off, 16);

        const float rinv = 1.0f / s;
        #pragma unroll
        for (int nt = 0; nt < 16; ++nt) acc[nt][r] *= rinv;
    }

    // ---- D-layout -> LDS (row major f16) so we can reload P in A-layout
    half_t* Pw = Plds + wave * 16 * NB;
    #pragma unroll
    for (int nt = 0; nt < 16; ++nt) {
        const int col = nt * 16 + l16;
        #pragma unroll
        for (int r = 0; r < 8; ++r) {
            const int row = r + hi * 8;        // C/D layout: upper half holds M+8
            Pw[row * NB + col] = (half_t)acc[nt][r];
        }
    }
    // same-wave DS ops are in-order; data-return waits inserted by compiler

    // ---- out = P (16x256) . comp_V (256x128): 8 n-chunks x 8 d-tiles
    v8f oacc[8];
    #pragma unroll
    for (int dt = 0; dt < 8; ++dt) oacc[dt] = (v8f){};

    const half_t* cVb = cVT + (size_t)b * D_DIM * NB;
    #pragma unroll
    for (int kc = 0; kc < 8; ++kc) {                   // K(=n) chunk of 32
        v16h ap;
        const half_t* prow = Pw + l16 * NB + kc * 32 + hi * 16;
        *(v8h*)&ap        = *(const v8h*)(prow);       // K b..b+7   (16B LDS)
        *(((v8h*)&ap) + 1)= *(const v8h*)(prow + 8);   // K b+8..b+15
        #pragma unroll
        for (int dt = 0; dt < 8; ++dt) {
            const int dcol = dt * 16 + l16;
            v16h bv = *(const v16h*)(cVb + (size_t)dcol * NB + kc * 32 + hi * 16);
            oacc[dt] = __builtin_amdgcn_wmma_f32_16x16x32_f16(false, ap,
                                                              false, bv,
                                                              (short)0, oacc[dt],
                                                              false, false);
        }
    }

    // ---- gate multiply + store
    const float* gb = gate + (size_t)b * T_LEN;
    float g[8];
    #pragma unroll
    for (int r = 0; r < 8; ++r) g[r] = gb[t0 + hi * 8 + r];

    float* ob = out + (size_t)b * T_LEN * D_DIM;
    #pragma unroll
    for (int r = 0; r < 8; ++r) {
        const size_t row = (size_t)(t0 + hi * 8 + r);
        #pragma unroll
        for (int dt = 0; dt < 8; ++dt)
            ob[row * D_DIM + dt * 16 + l16] = oacc[dt][r] * g[r];
    }
}

// ---------------------------------------------------------------------------
extern "C" void kernel_launch(void* const* d_in, const int* in_sizes, int n_in,
                              void* d_out, int out_size, void* d_ws, size_t ws_size,
                              hipStream_t stream)
{
    const float* Q = (const float*)d_in[0];
    const float* K = (const float*)d_in[1];
    const float* V = (const float*)d_in[2];
    const float* G = (const float*)d_in[3];

    const int B = in_sizes[0] / (T_LEN * D_DIM);   // 8 for reference shapes

    // workspace: comp_K16 [B][NB][D] then comp_VT16 [B][D][NB]  (f16, ~1MB total)
    half_t* cK  = (half_t*)d_ws;
    half_t* cVT = cK + (size_t)B * NB * D_DIM;

    nsa_compress<<<B * NB, 128, 0, stream>>>(K, V, cK, cVT);
    nsa_attn<<<B * (T_LEN / MROWS), 128, 0, stream>>>(Q, cK, cVT, G, (float*)d_out);
}
